// GraphMatchingNetwork_10264971837817
// MI455X (gfx1250) — matrix operand order repaired and checked by reading, not compiled, not run on hardware
//
#include <hip/hip_runtime.h>
#include <hip/hip_bf16.h>
#include <math.h>

#define NHEAD 8
#define DIM   768
#define DHEAD 96
#define NLAY  2
#define NGR   64
#define NN    4096
#define EPSV  1e-5f

typedef __bf16 bf16;
typedef __attribute__((ext_vector_type(16))) __bf16 v16bf;
typedef __attribute__((ext_vector_type(8)))  __bf16 v8bf;
typedef __attribute__((ext_vector_type(8)))  float  v8f;
typedef __attribute__((ext_vector_type(4)))  unsigned int v4u;
typedef __attribute__((ext_vector_type(8)))  int v8i;
typedef __attribute__((ext_vector_type(4)))  int v4i;

union FragB { v16bf v; v8bf h[2]; };

__device__ __forceinline__ v8f wmma_bf16(v16bf a, v16bf b, v8f c) {
  // D = A(16x32 bf16) x B(32x16 bf16) + C(16x16 f32)
  return __builtin_amdgcn_wmma_f32_16x16x32_bf16(false, a, false, b, (short)0, c,
                                                 false, false);
}

// TDM: load a 2D tile (tile_dim0=64 elems x tile_dim1=rows, 2B elements) of a
// row-major tensor (dim0=768 elems/row, dim1=tensor_rows, stride0=768) into
// LDS, padding each 128B row with 16B (pad_interval=4 -> 32 DWORDs,
// pad_amount=3 -> 4 DWORDs) so the LDS row stride is 144B: 16B-aligned for the
// ds_load_b128 fragment reads and bank-conflict-free (36-DWORD stride).
__device__ __forceinline__ void tdm_load_tile(const bf16* gptr, unsigned lds_addr,
                                              int rows, int tensor_rows) {
  unsigned long long ga = (unsigned long long)(size_t)gptr;
  v4u g0 = { 1u,                                     // count=1, user mode
             lds_addr,                               // D#.lds_addr
             (unsigned)(ga & 0xFFFFFFFFu),           // global_addr[31:0]
             (unsigned)((ga >> 32) & 0x01FFFFFFu) |  // global_addr[56:32]
                 (2u << 30) };                       // type=2 (image)
  v8i g1 = { (int)((1u << 16) |                      // data_size = 2 bytes
                   (1u << 20) |                      // pad_enable
                   (4u << 22) |                      // pad_interval: 32 DWORDs
                   (3u << 25)),                      // pad_amount: 4 DWORDs
             (int)(768u << 16),                      // tensor_dim0[15:0] = 768
             (int)(((unsigned)tensor_rows & 0xFFFFu) << 16),  // dim1[15:0]
             (int)((64u << 16) | ((unsigned)tensor_rows >> 16)),  // tile_dim0=64
             rows,                                   // tile_dim1, tile_dim2=0
             768,                                    // tensor_dim0_stride[31:0]
             0, 0 };
  v4i z4 = {0, 0, 0, 0};
#if defined(__clang_major__) && (__clang_major__ >= 23)
  v8i z8 = {0, 0, 0, 0, 0, 0, 0, 0};
  __builtin_amdgcn_tensor_load_to_lds(g0, g1, z4, z4, z8, 0);
#else
  __builtin_amdgcn_tensor_load_to_lds(g0, g1, z4, z4, 0);
#endif
}

__device__ __forceinline__ int lower_bound_i(const int* __restrict__ a, int n, int v) {
  int lo = 0, hi = n;
  while (lo < hi) { int mid = (lo + hi) >> 1; if (a[mid] < v) lo = mid + 1; else hi = mid; }
  return lo;
}

// ---------------------------------------------------------------- casts
__global__ void init_state(const float* __restrict__ src, float* __restrict__ f,
                           bf16* __restrict__ b, int n) {
  int i = blockIdx.x * blockDim.x + threadIdx.x;
  int stride = gridDim.x * blockDim.x;
  for (; i < n; i += stride) { float v = src[i]; f[i] = v; b[i] = (bf16)v; }
}

// Transpose-cast weights: dst[l][n][k] = (bf16)src[l][k][n]. Tiled via LDS so
// both global sides are coalesced. grid (DIM/32, DIM/32, NLAY), block (32,8).
__global__ __launch_bounds__(256)
void transpose_cast_w(const float* __restrict__ src, bf16* __restrict__ dst) {
  __shared__ float t[32][33];
  const int l = blockIdx.z;
  const int kb = blockIdx.x * 32, nb = blockIdx.y * 32;
  const int tx = threadIdx.x, ty = threadIdx.y;
  const size_t base = (size_t)l * DIM * DIM;
#pragma unroll
  for (int i = ty; i < 32; i += 8)
    t[i][tx] = src[base + (size_t)(kb + i) * DIM + nb + tx];
  __syncthreads();
#pragma unroll
  for (int i = ty; i < 32; i += 8)
    dst[base + (size_t)(nb + i) * DIM + kb + tx] = (bf16)t[tx][i];
}

// ---------------------------------------------------------------- GEMM
// Y[4096,768] = X[4096,768] @ W[768,768] + bias (+resid), with W supplied
// pre-transposed (Wt[n][k]). 128x64 block tile, 8 waves x 32x32 wave tiles.
// K stepped by 64 with double-buffered LDS stages; BOTH tiles fetched by the
// Tensor Data Mover (wave 0 issues 2 descriptors, waits TENSORcnt, barrier
// publishes). Inner loop is pure ds_load_b128 + v_wmma. One barrier per stage.
__global__ __launch_bounds__(256)
void gemm_bf16(const bf16* __restrict__ X, const bf16* __restrict__ Wt,
               const float* __restrict__ bias, const float* __restrict__ resid,
               float* __restrict__ outf, bf16* __restrict__ outb) {
  __shared__ __align__(16) bf16 As[2][128 * 72];   // 128 rows x 64 k (TDM pad -> 72)
  __shared__ __align__(16) bf16 Bst[2][64 * 72];   // 64 cols x 64 k (TDM pad -> 72)

  const int tid = threadIdx.x;
  const int lane = tid & 31, wave = tid >> 5;
  const int h = lane >> 4, ln15 = lane & 15;
  const int wm = wave >> 1, wn = wave & 1;
  const int m0 = blockIdx.x * 128;
  const int n0 = blockIdx.y * 64;

  v8f c[2][2];
  const v8f vz = {0.f, 0.f, 0.f, 0.f, 0.f, 0.f, 0.f, 0.f};
  c[0][0] = vz; c[0][1] = vz; c[1][0] = vz; c[1][1] = vz;

  auto load_stage = [&](int k0, int bufi) {
    if (wave == 0) {
      tdm_load_tile(X + (size_t)m0 * DIM + k0,
                    (unsigned)(size_t)&As[bufi][0], 128, NN);
      tdm_load_tile(Wt + (size_t)n0 * DIM + k0,
                    (unsigned)(size_t)&Bst[bufi][0], 64, DIM);
    }
  };

  load_stage(0, 0);
  if (wave == 0) __builtin_amdgcn_s_wait_tensorcnt(0);
  __syncthreads();

  int buf = 0;
  const int nstage = DIM / 64;
  for (int s = 0; s < nstage; ++s) {
    if (s + 1 < nstage) load_stage((s + 1) * 64, buf ^ 1);

#pragma unroll
    for (int ks = 0; ks < 2; ++ks) {
      FragB a[2], b[2];
#pragma unroll
      for (int i = 0; i < 2; ++i) {
        const bf16* p = &As[buf][(wm * 32 + i * 16 + ln15) * 72 + ks * 32];
        a[i].h[0] = *(const v8bf*)(p + h * 8);
        a[i].h[1] = *(const v8bf*)(p + 16 + h * 8);
      }
#pragma unroll
      for (int j = 0; j < 2; ++j) {
        const bf16* p = &Bst[buf][(wn * 32 + j * 16 + ln15) * 72 + ks * 32];
        b[j].h[0] = *(const v8bf*)(p + h * 8);
        b[j].h[1] = *(const v8bf*)(p + 16 + h * 8);
      }
#pragma unroll
      for (int i = 0; i < 2; ++i)
#pragma unroll
        for (int j = 0; j < 2; ++j)
          c[i][j] = wmma_bf16(a[i].v, b[j].v, c[i][j]);
    }

    if (s + 1 < nstage) {
      if (wave == 0) __builtin_amdgcn_s_wait_tensorcnt(0);
      __syncthreads();
    }
    buf ^= 1;
  }

#pragma unroll
  for (int i = 0; i < 2; ++i)
#pragma unroll
    for (int j = 0; j < 2; ++j)
#pragma unroll
      for (int r = 0; r < 8; ++r) {
        int row = m0 + wm * 32 + i * 16 + r + 8 * h;
        int col = n0 + wn * 32 + j * 16 + ln15;
        float v = c[i][j][r] + bias[col];
        if (resid) v += resid[(size_t)row * DIM + col];
        if (outf) outf[(size_t)row * DIM + col] = v;
        if (outb) outb[(size_t)row * DIM + col] = (bf16)v;
      }
}

// ------------------------------------------------------------ attention
// Block: one head x 64 query rows (4 waves x 16 rows). Flash-style online
// softmax; key range limited to matching graph ids via binary search.
__global__ __launch_bounds__(128)
void attn_kernel(const bf16* __restrict__ Q, const bf16* __restrict__ K,
                 const bf16* __restrict__ V, const int* __restrict__ batq,
                 const int* __restrict__ batk, bf16* __restrict__ O) {
  __shared__ __align__(16) bf16 Ks[32 * 104];     // 32 keys x 96 dh (pad 104)
  __shared__ __align__(16) bf16 Vt[96 * 40];      // 96 dh x 32 keys (pad 40)
  __shared__ __align__(16) bf16 Pl[4 * 16 * 32];  // per-wave P transpose scratch
  __shared__ int rng[2];

  const int tid = threadIdx.x;
  const int lane = tid & 31, wave = tid >> 5;
  const int h = lane >> 4, ln15 = lane & 15;
  const int hd = blockIdx.y;
  const int qlo = blockIdx.x * 64;
  const int q0 = qlo + wave * 16;

  if (tid == 0) {
    int gmin = batq[qlo], gmax = batq[qlo + 63];
    int lo = lower_bound_i(batk, NN, gmin);
    int hi = lower_bound_i(batk, NN, gmax + 1);
    rng[0] = lo & ~31;
    int up = (hi + 31) & ~31;
    rng[1] = up > NN ? NN : up;
  }

  // Q fragments (A layout), fixed over the key loop.
  FragB aq[3];
  const bf16* qrow = Q + (size_t)(q0 + ln15) * DIM + hd * DHEAD;
#pragma unroll
  for (int b = 0; b < 3; ++b) {
    aq[b].h[0] = *(const v8bf*)(qrow + 32 * b + h * 8);
    aq[b].h[1] = *(const v8bf*)(qrow + 32 * b + 16 + h * 8);
  }
  int bq_r[8];
#pragma unroll
  for (int r = 0; r < 8; ++r) bq_r[r] = batq[q0 + r + 8 * h];

  v8f o[6];
  const v8f vz = {0.f, 0.f, 0.f, 0.f, 0.f, 0.f, 0.f, 0.f};
#pragma unroll
  for (int n = 0; n < 6; ++n) o[n] = vz;
  float m_run[8], l_run[8];
#pragma unroll
  for (int r = 0; r < 8; ++r) { m_run[r] = -1.0e30f; l_run[r] = 0.f; }

  __syncthreads();
  const int mstart = rng[0], mend = rng[1];
  const float scale = 0.10206207262f;  // 1/sqrt(96)

  const int crow = tid >> 2;           // 0..31 key row
  const int ccol = (tid & 3) * 24;     // 0,24,48,72 dh column

  for (int m0 = mstart; m0 < mend; m0 += 32) {
    if (m0 + 32 < mend) {  // prefetch next K/V tile -> global_prefetch_b8
      __builtin_prefetch(K + (size_t)(m0 + 32 + crow) * DIM + hd * DHEAD + ccol, 0, 1);
      __builtin_prefetch(V + (size_t)(m0 + 32 + crow) * DIM + hd * DHEAD + ccol, 0, 1);
    }
    {  // cooperative K tile + transposed V tile load
      const bf16* kg = K + (size_t)(m0 + crow) * DIM + hd * DHEAD + ccol;
      uint4 k0v = *(const uint4*)(kg);
      uint4 k1v = *(const uint4*)(kg + 8);
      uint4 k2v = *(const uint4*)(kg + 16);
      bf16* kd = Ks + crow * 104 + ccol;
      *(uint4*)kd = k0v; *(uint4*)(kd + 8) = k1v; *(uint4*)(kd + 16) = k2v;
      const bf16* vg = V + (size_t)(m0 + crow) * DIM + hd * DHEAD + ccol;
#pragma unroll
      for (int j = 0; j < 24; ++j) Vt[(ccol + j) * 40 + crow] = vg[j];
    }
    __syncthreads();

    // S[16q, 32m] = Q K^T over dh=96 (3 k-steps, 2 m-subtiles)
    v8f s[2]; s[0] = vz; s[1] = vz;
#pragma unroll
    for (int b = 0; b < 3; ++b)
#pragma unroll
      for (int t = 0; t < 2; ++t) {
        FragB bk;
        const bf16* p = Ks + (t * 16 + ln15) * 104 + 32 * b + h * 8;
        bk.h[0] = *(const v8bf*)p;
        bk.h[1] = *(const v8bf*)(p + 16);
        s[t] = wmma_bf16(aq[b].v, bk.v, s[t]);
      }

    // mask + online softmax (rows span 16-lane halves)
    int mid0 = batk[m0 + ln15];
    int mid1 = batk[m0 + 16 + ln15];
    float p0a[8], p1a[8];
#pragma unroll
    for (int r = 0; r < 8; ++r) {
      bool ok0 = (bq_r[r] == mid0), ok1 = (bq_r[r] == mid1);
      float x0 = ok0 ? s[0][r] * scale : -3.0e30f;
      float x1 = ok1 ? s[1][r] * scale : -3.0e30f;
      float mx = fmaxf(x0, x1);
      mx = fmaxf(mx, __shfl_xor(mx, 1, 32));
      mx = fmaxf(mx, __shfl_xor(mx, 2, 32));
      mx = fmaxf(mx, __shfl_xor(mx, 4, 32));
      mx = fmaxf(mx, __shfl_xor(mx, 8, 32));
      float nm = fmaxf(m_run[r], mx);
      float corr = __expf(m_run[r] - nm);
      float p0 = ok0 ? __expf(x0 - nm) : 0.f;
      float p1 = ok1 ? __expf(x1 - nm) : 0.f;
      float ps = p0 + p1;
      ps += __shfl_xor(ps, 1, 32);
      ps += __shfl_xor(ps, 2, 32);
      ps += __shfl_xor(ps, 4, 32);
      ps += __shfl_xor(ps, 8, 32);
      l_run[r] = l_run[r] * corr + ps;
      m_run[r] = nm;
#pragma unroll
      for (int n = 0; n < 6; ++n) o[n][r] *= corr;
      p0a[r] = p0; p1a[r] = p1;
    }

    // P: C layout -> A layout via per-wave LDS scratch
    bf16* pw = Pl + wave * 16 * 32;
#pragma unroll
    for (int r = 0; r < 8; ++r) {
      pw[(r + 8 * h) * 32 + ln15]      = (bf16)p0a[r];
      pw[(r + 8 * h) * 32 + 16 + ln15] = (bf16)p1a[r];
    }
    __syncthreads();
    FragB ap;
    const bf16* pr = pw + ln15 * 32;
    ap.h[0] = *(const v8bf*)(pr + h * 8);
    ap.h[1] = *(const v8bf*)(pr + 16 + h * 8);
#pragma unroll
    for (int n = 0; n < 6; ++n) {
      FragB bv;
      const bf16* p = Vt + (n * 16 + ln15) * 40 + h * 8;
      bv.h[0] = *(const v8bf*)p;
      bv.h[1] = *(const v8bf*)(p + 16);
      o[n] = wmma_bf16(ap.v, bv.v, o[n]);
    }
    __syncthreads();
  }

#pragma unroll
  for (int n = 0; n < 6; ++n)
#pragma unroll
    for (int r = 0; r < 8; ++r) {
      int row = q0 + r + 8 * h;
      int col = hd * DHEAD + n * 16 + ln15;
      O[(size_t)row * DIM + col] = (bf16)(o[n][r] / l_run[r]);
    }
}

// ---------------------------------------------------------------- layernorm
__global__ __launch_bounds__(256)
void ln_kernel(const float* __restrict__ X, const float* __restrict__ g,
               const float* __restrict__ b, float* __restrict__ outf,
               bf16* __restrict__ outb) {
  __shared__ float red[256];
  const int row = blockIdx.x, tid = threadIdx.x;
  const float* x = X + (size_t)row * DIM;
  float v0 = x[tid], v1 = x[tid + 256], v2 = x[tid + 512];
  red[tid] = v0 + v1 + v2;
  __syncthreads();
  for (int off = 128; off; off >>= 1) { if (tid < off) red[tid] += red[tid + off]; __syncthreads(); }
  float mean = red[0] * (1.0f / DIM);
  __syncthreads();
  red[tid] = v0 * v0 + v1 * v1 + v2 * v2;
  __syncthreads();
  for (int off = 128; off; off >>= 1) { if (tid < off) red[tid] += red[tid + off]; __syncthreads(); }
  float var = red[0] * (1.0f / DIM) - mean * mean;
  float inv = rsqrtf(var + EPSV);
  float vv[3] = {v0, v1, v2};
#pragma unroll
  for (int i = 0; i < 3; ++i) {
    int d = tid + 256 * i;
    float y = (vv[i] - mean) * inv * g[d] + b[d];
    outf[(size_t)row * DIM + d] = y;
    outb[(size_t)row * DIM + d] = (bf16)y;
  }
}

// ---------------------------------------------------------------- aggregate
__global__ __launch_bounds__(256)
void agg_kernel(const float* __restrict__ Hf, const int* __restrict__ bat,
                float* __restrict__ G) {
  const int g = blockIdx.x;
  const int lo = lower_bound_i(bat, NN, g);
  const int hi = lower_bound_i(bat, NN, g + 1);
  const int cnt = hi - lo;
  for (int d = threadIdx.x; d < DIM; d += 256) {
    float s = 0.f, mx = -3.0e38f;
    for (int r = lo; r < hi; ++r) {
      float v = Hf[(size_t)r * DIM + d];
      s += v; mx = fmaxf(mx, v);
    }
    G[(size_t)g * DIM + d] = (cnt > 0) ? (s / (float)cnt + mx) : 0.f;
  }
}

// ---------------------------------------------------------------- MLP
__global__ __launch_bounds__(256)
void mlp1_kernel(const float* __restrict__ g1, const float* __restrict__ g2,
                 const float* __restrict__ W1, const float* __restrict__ b1,
                 float* __restrict__ hid) {
  __shared__ float sg[2 * DIM];
  const int g = blockIdx.x;
  for (int k = threadIdx.x; k < DIM; k += 256) {
    sg[k]       = g1[(size_t)g * DIM + k];
    sg[DIM + k] = g2[(size_t)g * DIM + k];
  }
  __syncthreads();
  for (int j = threadIdx.x; j < DIM; j += 256) {
    float acc = b1[j];
    for (int k = 0; k < 2 * DIM; ++k) acc += sg[k] * W1[(size_t)k * DIM + j];
    hid[(size_t)g * DIM + j] = fmaxf(acc, 0.f);
  }
}

__global__ __launch_bounds__(256)
void mlp2_kernel(const float* __restrict__ hid, const float* __restrict__ W2,
                 const float* __restrict__ b2, float* __restrict__ out) {
  __shared__ float red[256];
  const int g = blockIdx.x, tid = threadIdx.x;
  float acc = 0.f;
  for (int j = tid; j < DIM; j += 256) acc += hid[(size_t)g * DIM + j] * W2[j];
  red[tid] = acc;
  __syncthreads();
  for (int off = 128; off; off >>= 1) { if (tid < off) red[tid] += red[tid + off]; __syncthreads(); }
  if (tid == 0) {
    float z = red[0] + b2[0];
    out[g] = 1.f / (1.f + __expf(-z));
  }
}

// ---------------------------------------------------------------- launch
extern "C" void kernel_launch(void* const* d_in, const int* in_sizes, int n_in,
                              void* d_out, int out_size, void* d_ws, size_t ws_size,
                              hipStream_t stream) {
  (void)in_sizes; (void)n_in; (void)out_size; (void)ws_size;
  const float* h1  = (const float*)d_in[0];
  const float* h2  = (const float*)d_in[1];
  const int* batch1 = (const int*)d_in[2];
  const int* batch2 = (const int*)d_in[3];
  const float* Wq = (const float*)d_in[4];
  const float* bq = (const float*)d_in[5];
  const float* Wk = (const float*)d_in[6];
  const float* bk = (const float*)d_in[7];
  const float* Wv = (const float*)d_in[8];
  const float* bv = (const float*)d_in[9];
  const float* Wo = (const float*)d_in[10];
  const float* bo = (const float*)d_in[11];
  const float* ln_g = (const float*)d_in[12];
  const float* ln_b = (const float*)d_in[13];
  const float* mW1 = (const float*)d_in[14];
  const float* mb1 = (const float*)d_in[15];
  const float* mW2 = (const float*)d_in[16];
  const float* mb2 = (const float*)d_in[17];
  float* outp = (float*)d_out;

  char* ws = (char*)d_ws;
  size_t off = 0;
  auto alloc = [&](size_t bytes) -> void* {
    void* p = ws + off;
    off = (off + bytes + 255) & ~(size_t)255;
    return p;
  };
  const size_t ND  = (size_t)NN * DIM;
  const size_t WSZ = (size_t)NLAY * DIM * DIM;

  float* h1f  = (float*)alloc(ND * 4);
  float* h2f  = (float*)alloc(ND * 4);
  float* tmpf = (float*)alloc(ND * 4);
  bf16* h1b   = (bf16*)alloc(ND * 2);
  bf16* h2b   = (bf16*)alloc(ND * 2);
  bf16* qb    = (bf16*)alloc(ND * 2);
  bf16* kb    = (bf16*)alloc(ND * 2);
  bf16* vb    = (bf16*)alloc(ND * 2);
  bf16* attnb = (bf16*)alloc(ND * 2);
  bf16* Wqt   = (bf16*)alloc(WSZ * 2);   // pre-transposed bf16 weights [l][n][k]
  bf16* Wkt   = (bf16*)alloc(WSZ * 2);
  bf16* Wvt   = (bf16*)alloc(WSZ * 2);
  bf16* Wot   = (bf16*)alloc(WSZ * 2);
  float* g1   = (float*)alloc((size_t)NGR * DIM * 4);
  float* g2   = (float*)alloc((size_t)NGR * DIM * 4);
  float* hid  = (float*)alloc((size_t)NGR * DIM * 4);

  const dim3 tgrid(DIM / 32, DIM / 32, NLAY);
  const dim3 tblock(32, 8);
  transpose_cast_w<<<tgrid, tblock, 0, stream>>>(Wq, Wqt);
  transpose_cast_w<<<tgrid, tblock, 0, stream>>>(Wk, Wkt);
  transpose_cast_w<<<tgrid, tblock, 0, stream>>>(Wv, Wvt);
  transpose_cast_w<<<tgrid, tblock, 0, stream>>>(Wo, Wot);
  init_state<<<2048, 256, 0, stream>>>(h1, h1f, h1b, (int)ND);
  init_state<<<2048, 256, 0, stream>>>(h2, h2f, h2b, (int)ND);

  const dim3 ggrid(NN / 128, DIM / 64);
  const dim3 agrid(NN / 64, NHEAD);

  for (int i = 0; i < NLAY; ++i) {
    const bf16* Wq_i = Wqt + (size_t)i * DIM * DIM;
    const bf16* Wk_i = Wkt + (size_t)i * DIM * DIM;
    const bf16* Wv_i = Wvt + (size_t)i * DIM * DIM;
    const bf16* Wo_i = Wot + (size_t)i * DIM * DIM;
    const float* bq_i = bq + (size_t)i * DIM;
    const float* bk_i = bk + (size_t)i * DIM;
    const float* bv_i = bv + (size_t)i * DIM;
    const float* bo_i = bo + (size_t)i * DIM;

    // m12: queries h1, kv h2
    gemm_bf16<<<ggrid, 256, 0, stream>>>(h1b, Wq_i, bq_i, nullptr, nullptr, qb);
    gemm_bf16<<<ggrid, 256, 0, stream>>>(h2b, Wk_i, bk_i, nullptr, nullptr, kb);
    gemm_bf16<<<ggrid, 256, 0, stream>>>(h2b, Wv_i, bv_i, nullptr, nullptr, vb);
    attn_kernel<<<agrid, 128, 0, stream>>>(qb, kb, vb, batch1, batch2, attnb);
    gemm_bf16<<<ggrid, 256, 0, stream>>>(attnb, Wo_i, bo_i, h1f, tmpf, nullptr);
    ln_kernel<<<NN, 256, 0, stream>>>(tmpf, ln_g + (size_t)(2 * i) * DIM,
                                      ln_b + (size_t)(2 * i) * DIM, h1f, h1b);

    // m21: queries h2, kv h1 (updated)
    gemm_bf16<<<ggrid, 256, 0, stream>>>(h2b, Wq_i, bq_i, nullptr, nullptr, qb);
    gemm_bf16<<<ggrid, 256, 0, stream>>>(h1b, Wk_i, bk_i, nullptr, nullptr, kb);
    gemm_bf16<<<ggrid, 256, 0, stream>>>(h1b, Wv_i, bv_i, nullptr, nullptr, vb);
    attn_kernel<<<agrid, 128, 0, stream>>>(qb, kb, vb, batch2, batch1, attnb);
    gemm_bf16<<<ggrid, 256, 0, stream>>>(attnb, Wo_i, bo_i, h2f, tmpf, nullptr);
    ln_kernel<<<NN, 256, 0, stream>>>(tmpf, ln_g + (size_t)(2 * i + 1) * DIM,
                                      ln_b + (size_t)(2 * i + 1) * DIM, h2f, h2b);
  }

  agg_kernel<<<NGR, 256, 0, stream>>>(h1f, batch1, g1);
  agg_kernel<<<NGR, 256, 0, stream>>>(h2f, batch2, g2);
  mlp1_kernel<<<NGR, 256, 0, stream>>>(g1, g2, mW1, mb1, hid);
  mlp2_kernel<<<NGR, 256, 0, stream>>>(hid, mW2, mb2, outp);
}
